// RMHALayer_81922206204068
// MI455X (gfx1250) — compile-verified, matrix operand order checked by reading, not run on hardware
//
#include <hip/hip_runtime.h>
#include <hip/hip_bf16.h>

// ---------------------------------------------------------------------------
// Problem constants (match reference)
// ---------------------------------------------------------------------------
#define Bb   4
#define Nn   512
#define Hh   512
#define NH   8
#define HD   64
#define DE   64
#define DFF  2048
#define Mrows (Bb * Nn)   // 2048

typedef __bf16 bf16_t;
typedef __attribute__((ext_vector_type(16))) __bf16 v16bf;
typedef __attribute__((ext_vector_type(8)))  float  v8f;
typedef __attribute__((ext_vector_type(4)))  unsigned int v4u;

union FragU { v16bf v; v4u u[2]; };

// Load a 16x32 bf16 WMMA fragment (A layout; B operand uses the same layout
// with the matrix stored N-major / K-contiguous, i.e. computes X @ W^T).
// lane L<16 : row L,    K = [k0..k0+7] and [k0+16..k0+23]
// lane L>=16: row L-16, K = [k0+8..k0+15] and [k0+24..k0+31]
__device__ __forceinline__ v16bf load_frag(const bf16_t* base, int ld, int k0) {
    int lane = threadIdx.x & 31;
    const bf16_t* p = base + (size_t)(lane & 15) * ld + k0 + ((lane >> 4) << 3);
    FragU f;
    f.u[0] = *reinterpret_cast<const v4u*>(p);
    f.u[1] = *reinterpret_cast<const v4u*>(p + 16);
    return f.v;
}

__device__ __forceinline__ v8f wmma_bf16(v16bf a, v16bf b, v8f c) {
    return __builtin_amdgcn_wmma_f32_16x16x32_bf16(
        false, a, false, b, (short)0, c, false, false);
}

// ---------------------------------------------------------------------------
// f32 -> bf16 conversion
// ---------------------------------------------------------------------------
__global__ void cvt_kernel(const float* __restrict__ in, bf16_t* __restrict__ out, int n) {
    int i = blockIdx.x * 256 + threadIdx.x;
    if (i < n) out[i] = (bf16_t)in[i];
}

// ---------------------------------------------------------------------------
// Distance-embedding means.
//  meanq[b,n,:] = mean_m emb[idx[b,n,m]]   (gridIdx.y == 0)
//  meank[b,p,:] = mean_n emb[idx[b,n,p]]   (gridIdx.y == 1)
// block = 64 threads (one per DE element)
// ---------------------------------------------------------------------------
__global__ void meanemb_kernel(const float* __restrict__ dm,
                               const float* __restrict__ emb,
                               bf16_t* __restrict__ meanq,
                               bf16_t* __restrict__ meank) {
    int d  = threadIdx.x;
    int bn = blockIdx.x;                // b*Nn + n
    int b  = bn >> 9;
    int n  = bn & (Nn - 1);
    float acc = 0.f;
    if (blockIdx.y == 0) {
        const float* drow = dm + (size_t)bn * Nn;
        for (int m = 0; m < Nn; ++m) {
            int idx = (int)drow[m];
            idx = min(max(idx, 0), 200);
            acc += emb[idx * DE + d];
        }
        meanq[(size_t)bn * DE + d] = (bf16_t)(acc * (1.f / (float)Nn));
    } else {
        const float* dcol = dm + ((size_t)b * Nn) * Nn + n;   // vary row index
        for (int m = 0; m < Nn; ++m) {
            int idx = (int)dcol[(size_t)m * Nn];
            idx = min(max(idx, 0), 200);
            acc += emb[idx * DE + d];
        }
        meank[(size_t)bn * DE + d] = (bf16_t)(acc * (1.f / (float)Nn));
    }
}

// ---------------------------------------------------------------------------
// Generic bf16 WMMA GEMM:  C(M,N) = act( A(M,K) @ W(N,K)^T + bias [+ res] )
// Register-blocked: one wave computes a 32x64 output block (2x4 tiles of
// 16x16), reusing 2 A-fragments and 4 B-fragments per K-step:
// 8 WMMAs per 6 fragment loads. grid = (N/64, M/32), block = 32.
// ---------------------------------------------------------------------------
#define ACT_NONE 0
#define ACT_RELU 1

__global__ void gemm_kernel(const bf16_t* __restrict__ A,
                            const bf16_t* __restrict__ W,
                            const float*  __restrict__ bias,
                            const float*  __restrict__ res,
                            float*        __restrict__ outF,
                            bf16_t*       __restrict__ outB,
                            int N, int K, int act) {
    int n0 = blockIdx.x * 64;
    int m0 = blockIdx.y * 32;
    const bf16_t* Ab0 = A + (size_t)m0 * K;
    const bf16_t* Ab1 = Ab0 + (size_t)16 * K;
    const bf16_t* Wb0 = W + (size_t)n0 * K;
    const bf16_t* Wb1 = Wb0 + (size_t)16 * K;
    const bf16_t* Wb2 = Wb0 + (size_t)32 * K;
    const bf16_t* Wb3 = Wb0 + (size_t)48 * K;

    v8f acc00 = {}, acc01 = {}, acc02 = {}, acc03 = {};
    v8f acc10 = {}, acc11 = {}, acc12 = {}, acc13 = {};

    for (int k0 = 0; k0 < K; k0 += 32) {
        v16bf a0 = load_frag(Ab0, K, k0);
        v16bf a1 = load_frag(Ab1, K, k0);
        v16bf b0 = load_frag(Wb0, K, k0);
        v16bf b1 = load_frag(Wb1, K, k0);
        v16bf b2 = load_frag(Wb2, K, k0);
        v16bf b3 = load_frag(Wb3, K, k0);
        acc00 = wmma_bf16(a0, b0, acc00);
        acc01 = wmma_bf16(a0, b1, acc01);
        acc02 = wmma_bf16(a0, b2, acc02);
        acc03 = wmma_bf16(a0, b3, acc03);
        acc10 = wmma_bf16(a1, b0, acc10);
        acc11 = wmma_bf16(a1, b1, acc11);
        acc12 = wmma_bf16(a1, b2, acc12);
        acc13 = wmma_bf16(a1, b3, acc13);
    }

    int lane  = threadIdx.x & 31;
    int ncol  = lane & 15;
    int mbase = (lane >> 4) << 3;

    v8f* accs[2][4] = { { &acc00, &acc01, &acc02, &acc03 },
                        { &acc10, &acc11, &acc12, &acc13 } };
#pragma unroll
    for (int ti = 0; ti < 2; ++ti) {
#pragma unroll
        for (int tj = 0; tj < 4; ++tj) {
            v8f a = *accs[ti][tj];
            int nb = n0 + tj * 16 + ncol;
            float bv = bias ? bias[nb] : 0.f;
#pragma unroll
            for (int i = 0; i < 8; ++i) {
                size_t m = (size_t)(m0 + ti * 16 + mbase + i);
                size_t o = m * N + nb;
                float v = a[i] + bv;
                if (res) v += res[o];
                if (act == ACT_RELU) v = fmaxf(v, 0.f);
                if (outF) outF[o] = v;
                if (outB) outB[o] = (bf16_t)v;
            }
        }
    }
}

// ---------------------------------------------------------------------------
// V transpose: Vt[(b*NH+h)*HD + d][j] = V[b*Nn + j][h*HD + d]
// ---------------------------------------------------------------------------
__global__ void vt_kernel(const bf16_t* __restrict__ V, bf16_t* __restrict__ Vt) {
    int i = blockIdx.x * 256 + threadIdx.x;      // over Bb*NH*HD*Nn = 1M
    int j  = i & (Nn - 1);
    int r  = i >> 9;
    int d  = r & (HD - 1);
    int bh = r >> 6;
    int h  = bh & (NH - 1);
    int b  = bh >> 3;
    Vt[i] = V[((size_t)(b * Nn + j)) * Hh + h * HD + d];
}

// ---------------------------------------------------------------------------
// Fused flash attention. One wave per (16 Q rows, head, batch).
// grid = (Nn/16, NH, Bb), block = 32.
// ao[b,m,h*HD+d] = softmax(mask( Q K^T /8 )) @ V
// ---------------------------------------------------------------------------
__global__ void attn_kernel(const bf16_t* __restrict__ Qb,
                            const bf16_t* __restrict__ Kb,
                            const bf16_t* __restrict__ Vt,
                            const float*  __restrict__ mask,
                            bf16_t*       __restrict__ ao) {
    __shared__ __align__(16) bf16_t pl[16 * 32];

    int m0 = blockIdx.x * 16;
    int h  = blockIdx.y;
    int b  = blockIdx.z;
    int lane  = threadIdx.x & 31;
    int ncol  = lane & 15;
    int mbase = (lane >> 4) << 3;

    const bf16_t* qbase = Qb + ((size_t)(b * Nn + m0)) * Hh + h * HD;
    v16bf qa0 = load_frag(qbase, Hh, 0);
    v16bf qa1 = load_frag(qbase, Hh, 32);

    v8f o0 = {}, o1 = {}, o2 = {}, o3 = {};
    float mrow[8], lrow[8];
#pragma unroll
    for (int i = 0; i < 8; ++i) { mrow[i] = -1e30f; lrow[i] = 0.f; }

    const float scale = 0.125f;   // 1/sqrt(64)
    const size_t mask_row0 = (size_t)(b * Nn + m0 + mbase) * Nn;

    for (int j0 = 0; j0 < Nn; j0 += 32) {
        // ---- S tile (16 x 32) = Q(16x64) @ K^T ----
        const bf16_t* kb0 = Kb + ((size_t)(b * Nn + j0)) * Hh + h * HD;
        const bf16_t* kb1 = kb0 + (size_t)16 * Hh;
        v8f s0 = {}, s1 = {};
        s0 = wmma_bf16(qa0, load_frag(kb0, Hh, 0), s0);
        s0 = wmma_bf16(qa1, load_frag(kb0, Hh, 32), s0);
        s1 = wmma_bf16(qa0, load_frag(kb1, Hh, 0), s1);
        s1 = wmma_bf16(qa1, load_frag(kb1, Hh, 32), s1);

        // ---- masked online softmax ----
        float v0[8], v1[8], mk0[8], mk1[8], tmax[8];
#pragma unroll
        for (int i = 0; i < 8; ++i) {
            mk0[i] = mask[mask_row0 + (size_t)i * Nn + j0 + ncol];
            mk1[i] = mask[mask_row0 + (size_t)i * Nn + j0 + 16 + ncol];
            float a0 = s0[i] * scale; if (mk0[i] == 0.f) a0 = -1e30f;
            float a1 = s1[i] * scale; if (mk1[i] == 0.f) a1 = -1e30f;
            v0[i] = a0; v1[i] = a1;
            float t = fmaxf(a0, a1);
            t = fmaxf(t, __shfl_xor(t, 1, 32));
            t = fmaxf(t, __shfl_xor(t, 2, 32));
            t = fmaxf(t, __shfl_xor(t, 4, 32));
            t = fmaxf(t, __shfl_xor(t, 8, 32));
            tmax[i] = t;
        }
#pragma unroll
        for (int i = 0; i < 8; ++i) {
            float mnew = fmaxf(mrow[i], tmax[i]);
            float corr = __expf(mrow[i] - mnew);
            mrow[i] = mnew;
            float e0 = (mk0[i] == 0.f) ? 0.f : __expf(v0[i] - mnew);
            float e1 = (mk1[i] == 0.f) ? 0.f : __expf(v1[i] - mnew);
            float rs = e0 + e1;
            rs += __shfl_xor(rs, 1, 32);
            rs += __shfl_xor(rs, 2, 32);
            rs += __shfl_xor(rs, 4, 32);
            rs += __shfl_xor(rs, 8, 32);
            lrow[i] = lrow[i] * corr + rs;
            o0[i] *= corr; o1[i] *= corr; o2[i] *= corr; o3[i] *= corr;
            pl[(mbase + i) * 32 + ncol]      = (bf16_t)e0;
            pl[(mbase + i) * 32 + 16 + ncol] = (bf16_t)e1;
        }
        __syncthreads();

        // ---- O += P(16x32) @ V(32x64) ----
        v16bf pf = load_frag(pl, 32, 0);
        const bf16_t* vb = Vt + ((size_t)((b * NH + h) * HD)) * Nn + j0;
        o0 = wmma_bf16(pf, load_frag(vb,                      Nn, 0), o0);
        o1 = wmma_bf16(pf, load_frag(vb + (size_t)16 * Nn,    Nn, 0), o1);
        o2 = wmma_bf16(pf, load_frag(vb + (size_t)32 * Nn,    Nn, 0), o2);
        o3 = wmma_bf16(pf, load_frag(vb + (size_t)48 * Nn,    Nn, 0), o3);
        __syncthreads();
    }

    // ---- normalize + store ----
    size_t robase = (size_t)(b * Nn + m0 + mbase);
#pragma unroll
    for (int i = 0; i < 8; ++i) {
        float inv = 1.f / lrow[i];
        size_t o = (robase + i) * Hh + h * HD + ncol;
        ao[o]      = (bf16_t)(o0[i] * inv);
        ao[o + 16] = (bf16_t)(o1[i] * inv);
        ao[o + 32] = (bf16_t)(o2[i] * inv);
        ao[o + 48] = (bf16_t)(o3[i] * inv);
    }
}

// ---------------------------------------------------------------------------
// GRU elementwise: u = (1-z)*n + z*msg  from gx, gh (M x 1536)
// ---------------------------------------------------------------------------
__global__ void gru_kernel(const float* __restrict__ gx,
                           const float* __restrict__ gh,
                           const float* __restrict__ msg,
                           float* __restrict__ uF, bf16_t* __restrict__ uB) {
    int idx = blockIdx.x * 256 + threadIdx.x;    // over Mrows*Hh
    int row = idx >> 9;
    int col = idx & (Hh - 1);
    size_t base = (size_t)row * (3 * Hh);
    float xr = gx[base + col], xz = gx[base + Hh + col], xn = gx[base + 2 * Hh + col];
    float hr = gh[base + col], hz = gh[base + Hh + col], hn = gh[base + 2 * Hh + col];
    float r = 1.f / (1.f + __expf(-(xr + hr)));
    float z = 1.f / (1.f + __expf(-(xz + hz)));
    float n = tanhf(xn + r * hn);
    float m = msg[idx];
    float u = (1.f - z) * n + z * m;
    uF[idx] = u;
    uB[idx] = (bf16_t)u;
}

// ---------------------------------------------------------------------------
// LayerNorm over last dim (512). block = 256 (2 elems/thread), grid = Mrows.
// ---------------------------------------------------------------------------
__global__ void ln_kernel(const float* __restrict__ y,
                          const float* __restrict__ g2,
                          const float* __restrict__ beta2,
                          float* __restrict__ out) {
    __shared__ float sred[8];
    __shared__ float smu, svar;
    int row = blockIdx.x;
    int t = threadIdx.x;
    const float* yr = y + (size_t)row * Hh;
    float a = yr[t], bq = yr[t + 256];
    float s = a + bq;
#pragma unroll
    for (int o = 16; o >= 1; o >>= 1) s += __shfl_xor(s, o, 32);
    if ((t & 31) == 0) sred[t >> 5] = s;
    __syncthreads();
    if (t == 0) {
        float tot = 0.f;
        for (int i = 0; i < 8; ++i) tot += sred[i];
        smu = tot * (1.f / (float)Hh);
    }
    __syncthreads();
    float mu = smu;
    float da = a - mu, db = bq - mu;
    float v = da * da + db * db;
#pragma unroll
    for (int o = 16; o >= 1; o >>= 1) v += __shfl_xor(v, o, 32);
    if ((t & 31) == 0) sred[t >> 5] = v;
    __syncthreads();
    if (t == 0) {
        float tot = 0.f;
        for (int i = 0; i < 8; ++i) tot += sred[i];
        svar = tot * (1.f / (float)Hh);
    }
    __syncthreads();
    float inv = rsqrtf(svar + 1e-5f);
    out[(size_t)row * Hh + t]       = da * inv * g2[t] + beta2[t];
    out[(size_t)row * Hh + t + 256] = db * inv * g2[t + 256] + beta2[t + 256];
}

// ---------------------------------------------------------------------------
// Host launcher
// ---------------------------------------------------------------------------
static inline void gemm_launch(hipStream_t s, const bf16_t* A, const bf16_t* W,
                               const float* bias, const float* res,
                               float* outF, bf16_t* outB, int M, int N, int K, int act) {
    dim3 g(N / 64, M / 32);
    gemm_kernel<<<g, 32, 0, s>>>(A, W, bias, res, outF, outB, N, K, act);
}

static inline void cvt_launch(hipStream_t s, const float* in, bf16_t* out, int n) {
    cvt_kernel<<<(n + 255) / 256, 256, 0, s>>>(in, out, n);
}

extern "C" void kernel_launch(void* const* d_in, const int* in_sizes, int n_in,
                              void* d_out, int out_size, void* d_ws, size_t ws_size,
                              hipStream_t stream) {
    const float* messages = (const float*)d_in[0];
    const float* distm    = (const float*)d_in[1];
    const float* mask     = (const float*)d_in[2];
    const float* embt     = (const float*)d_in[3];
    const float* Wdq = (const float*)d_in[4];  const float* bdq = (const float*)d_in[5];
    const float* Wdk = (const float*)d_in[6];  const float* bdk = (const float*)d_in[7];
    const float* Wq  = (const float*)d_in[8];  const float* bq  = (const float*)d_in[9];
    const float* Wk  = (const float*)d_in[10]; const float* bk  = (const float*)d_in[11];
    const float* Wv  = (const float*)d_in[12]; const float* bv  = (const float*)d_in[13];
    const float* Wo  = (const float*)d_in[14]; const float* bo  = (const float*)d_in[15];
    const float* Wih = (const float*)d_in[16]; const float* bih = (const float*)d_in[17];
    const float* Whh = (const float*)d_in[18]; const float* bhh = (const float*)d_in[19];
    const float* W1  = (const float*)d_in[20]; const float* b1  = (const float*)d_in[21];
    const float* W2  = (const float*)d_in[22]; const float* b2  = (const float*)d_in[23];
    const float* g2  = (const float*)d_in[24]; const float* be2 = (const float*)d_in[25];
    float* out = (float*)d_out;

    // workspace carve-up (256B aligned)
    size_t off = 0;
    char* wsb = (char*)d_ws;
    auto alloc = [&](size_t bytes) -> void* {
        void* p = wsb + off;
        off += (bytes + 255) & ~(size_t)255;
        return p;
    };
    const int M = Mrows;
    bf16_t* msg_bf  = (bf16_t*)alloc((size_t)M * Hh * 2);
    bf16_t* wdq_bf  = (bf16_t*)alloc((size_t)Hh * DE * 2);
    bf16_t* wdk_bf  = (bf16_t*)alloc((size_t)Hh * DE * 2);
    bf16_t* wq_bf   = (bf16_t*)alloc((size_t)Hh * Hh * 2);
    bf16_t* wk_bf   = (bf16_t*)alloc((size_t)Hh * Hh * 2);
    bf16_t* wv_bf   = (bf16_t*)alloc((size_t)Hh * Hh * 2);
    bf16_t* wo_bf   = (bf16_t*)alloc((size_t)Hh * Hh * 2);
    bf16_t* wih_bf  = (bf16_t*)alloc((size_t)3 * Hh * Hh * 2);
    bf16_t* whh_bf  = (bf16_t*)alloc((size_t)3 * Hh * Hh * 2);
    bf16_t* w1_bf   = (bf16_t*)alloc((size_t)DFF * Hh * 2);
    bf16_t* w2_bf   = (bf16_t*)alloc((size_t)Hh * DFF * 2);
    bf16_t* meanq   = (bf16_t*)alloc((size_t)M * DE * 2);
    bf16_t* meank   = (bf16_t*)alloc((size_t)M * DE * 2);
    bf16_t* xq_bf   = (bf16_t*)alloc((size_t)M * Hh * 2);
    bf16_t* xk_bf   = (bf16_t*)alloc((size_t)M * Hh * 2);
    bf16_t* Qb      = (bf16_t*)alloc((size_t)M * Hh * 2);
    bf16_t* Kbf     = (bf16_t*)alloc((size_t)M * Hh * 2);
    bf16_t* Vb      = (bf16_t*)alloc((size_t)M * Hh * 2);
    bf16_t* Vt      = (bf16_t*)alloc((size_t)M * Hh * 2);
    bf16_t* ao_bf   = (bf16_t*)alloc((size_t)M * Hh * 2);
    bf16_t* aoo_bf  = (bf16_t*)alloc((size_t)M * Hh * 2);
    float*  gx_f    = (float*)alloc((size_t)M * 3 * Hh * 4);
    float*  gh_f    = (float*)alloc((size_t)M * 3 * Hh * 4);
    float*  u_f     = (float*)alloc((size_t)M * Hh * 4);
    bf16_t* u_bf    = (bf16_t*)alloc((size_t)M * Hh * 2);
    bf16_t* f1_bf   = (bf16_t*)alloc((size_t)M * DFF * 2);
    float*  y_f     = (float*)alloc((size_t)M * Hh * 4);

    // 1) conversions
    cvt_launch(stream, messages, msg_bf, M * Hh);
    cvt_launch(stream, Wdq, wdq_bf, Hh * DE);
    cvt_launch(stream, Wdk, wdk_bf, Hh * DE);
    cvt_launch(stream, Wq,  wq_bf,  Hh * Hh);
    cvt_launch(stream, Wk,  wk_bf,  Hh * Hh);
    cvt_launch(stream, Wv,  wv_bf,  Hh * Hh);
    cvt_launch(stream, Wo,  wo_bf,  Hh * Hh);
    cvt_launch(stream, Wih, wih_bf, 3 * Hh * Hh);
    cvt_launch(stream, Whh, whh_bf, 3 * Hh * Hh);
    cvt_launch(stream, W1,  w1_bf,  DFF * Hh);
    cvt_launch(stream, W2,  w2_bf,  Hh * DFF);

    // 2) distance-embedding means
    meanemb_kernel<<<dim3(M, 2), 64, 0, stream>>>(distm, embt, meanq, meank);

    // 3) xq = messages + meanq @ Wdq^T + bdq ; xk likewise
    gemm_launch(stream, meanq, wdq_bf, bdq, messages, nullptr, xq_bf, M, Hh, DE, ACT_NONE);
    gemm_launch(stream, meank, wdk_bf, bdk, messages, nullptr, xk_bf, M, Hh, DE, ACT_NONE);

    // 4) Q, K, V projections
    gemm_launch(stream, xq_bf,  wq_bf, bq, nullptr, nullptr, Qb,  M, Hh, Hh, ACT_NONE);
    gemm_launch(stream, xk_bf,  wk_bf, bk, nullptr, nullptr, Kbf, M, Hh, Hh, ACT_NONE);
    gemm_launch(stream, msg_bf, wv_bf, bv, nullptr, nullptr, Vb,  M, Hh, Hh, ACT_NONE);
    vt_kernel<<<(M * Hh) / 256, 256, 0, stream>>>(Vb, Vt);

    // 5) fused attention
    attn_kernel<<<dim3(Nn / 16, NH, Bb), 32, 0, stream>>>(Qb, Kbf, Vt, mask, ao_bf);

    // 6) output projection
    gemm_launch(stream, ao_bf, wo_bf, bo, nullptr, nullptr, aoo_bf, M, Hh, Hh, ACT_NONE);

    // 7) GRU gates + elementwise
    gemm_launch(stream, aoo_bf, wih_bf, bih, nullptr, gx_f, nullptr, M, 3 * Hh, Hh, ACT_NONE);
    gemm_launch(stream, msg_bf, whh_bf, bhh, nullptr, gh_f, nullptr, M, 3 * Hh, Hh, ACT_NONE);
    gru_kernel<<<(M * Hh) / 256, 256, 0, stream>>>(gx_f, gh_f, messages, u_f, u_bf);

    // 8) FFN with fused residual
    gemm_launch(stream, u_bf,  w1_bf, b1, nullptr, nullptr, f1_bf, M, DFF, Hh, ACT_RELU);
    gemm_launch(stream, f1_bf, w2_bf, b2, u_f, y_f, nullptr, M, Hh, DFF, ACT_NONE);

    // 9) LayerNorm -> output
    ln_kernel<<<M, 256, 0, stream>>>(y_f, g2, be2, out);
}
